// LRU_71777493450999
// MI455X (gfx1250) — matrix operand order repaired and checked by reading, not compiled, not run on hardware
//
#include <hip/hip_runtime.h>
#include <hip/hip_bf16.h>
#include <stdint.h>

typedef __bf16 bf16;
typedef bf16  bf16x16 __attribute__((ext_vector_type(16)));
typedef float floatx8 __attribute__((ext_vector_type(8)));

constexpr int D_IN   = 512;
constexpr int D_H    = 1024;
constexpr int D_OUT  = 512;
constexpr int NBATCH = 8;
constexpr int SEQ    = 2048;
constexpr int M      = NBATCH * SEQ;   // 16384 rows (b*T+t)
constexpr int KCAT   = 2 * D_H + D_IN; // 2560 fused K for output GEMM

constexpr int LDT = 40;  // padded LDS row stride (bf16 units) for 32-wide K tiles

union FragU { bf16x16 v; uint4 q[2]; };

// A fragment (16x32 bf16, ISA 7.12.2): lane = 16*hi + m.
// halves j=0..7 -> K = hi*8 + j ; halves j=8..15 -> K = 16 + hi*8 + (j-8)
static __device__ __forceinline__ bf16x16 frag_a(const bf16* row_base, int lane) {
  const int m  = lane & 15;
  const int hi = lane >> 4;
  FragU f;
  f.q[0] = *(const uint4*)(row_base + m * LDT + hi * 8);
  f.q[1] = *(const uint4*)(row_base + m * LDT + 16 + hi * 8);
  return f.v;
}

// B fragment (32x16 bf16) from weights stored [n][k]: lane = 16*hi + n,
// halves j=0..7 -> K = hi*8 + j ; halves j=8..15 -> K = 16 + hi*8 + (j-8)
static __device__ __forceinline__ bf16x16 frag_b(const bf16* nrow_base, int lane) {
  const int n  = lane & 15;
  const int hi = lane >> 4;
  FragU f;
  f.q[0] = *(const uint4*)(nrow_base + n * LDT + hi * 8);
  f.q[1] = *(const uint4*)(nrow_base + n * LDT + 16 + hi * 8);
  return f.v;
}

// CDNA5 async HBM->LDS copy (16B per lane), tracked by ASYNCcnt.
// Generic-pointer low 32 bits == LDS byte offset (ISA aperture rule).
static __device__ __forceinline__ void async_b128(const bf16* lds_dst, const bf16* gsrc) {
  asm volatile("global_load_async_to_lds_b128 %0, %1, off"
               :: "v"((uint32_t)(uintptr_t)lds_dst),
                  "v"((uint64_t)(uintptr_t)gsrc)
               : "memory");
}
static __device__ __forceinline__ void wait_async0() {
  asm volatile("s_wait_asynccnt 0" ::: "memory");
}

// ---------------- prep kernels ----------------

__global__ __launch_bounds__(256) void lru_prep_params(
    const float* __restrict__ theta_log, const float* __restrict__ nu_log,
    float* __restrict__ lamRe, float* __restrict__ lamIm) {
  int h = blockIdx.x * 256 + threadIdx.x;
  if (h >= D_H) return;
  float mag = expf(-expf(nu_log[h]));
  float ang = expf(theta_log[h]);
  lamRe[h] = mag * cosf(ang);
  lamIm[h] = mag * sinf(ang);
}

__global__ __launch_bounds__(256) void lru_prep_b(
    const float* __restrict__ gamma_log,
    const float* __restrict__ Bre, const float* __restrict__ Bim,
    bf16* __restrict__ Wre, bf16* __restrict__ Wim) {
  int i = blockIdx.x * 256 + threadIdx.x;
  if (i >= D_H * D_IN) return;
  float g = gamma_log[i / D_IN];   // source uses gamma_log directly (no exp)
  Wre[i] = (bf16)(g * Bre[i]);
  Wim[i] = (bf16)(g * Bim[i]);
}

__global__ __launch_bounds__(256) void lru_prep_u(
    const float* __restrict__ u, bf16* __restrict__ ub) {
  int i = blockIdx.x * 256 + threadIdx.x;
  if (i >= M * D_IN) return;
  ub[i] = (bf16)u[i];
}

// Wcat[o, k] : k<1024 -> C_re[o,k] ; k<2048 -> -C_im[o,k-1024] ; else D[o,k-2048]
__global__ __launch_bounds__(256) void lru_prep_wcat(
    const float* __restrict__ Cre, const float* __restrict__ Cim,
    const float* __restrict__ Dm, bf16* __restrict__ Wcat) {
  int i = blockIdx.x * 256 + threadIdx.x;
  if (i >= D_OUT * KCAT) return;
  int n = i / KCAT, k = i % KCAT;
  float v;
  if (k < D_H)            v =  Cre[n * D_H + k];
  else if (k < 2 * D_H)   v = -Cim[n * D_H + (k - D_H)];
  else                    v =  Dm[n * D_IN + (k - 2 * D_H)];
  Wcat[i] = (bf16)v;
}

// ---------------- GEMM 1: Bu = u @ Bn^T (re & im share the A tile) ----------------
// Double-buffered LDS, tiles filled by async DMA overlapping WMMA compute.

__global__ __launch_bounds__(256) void lru_gemm_bu(
    const bf16* __restrict__ U,    // [M, D_IN]
    const bf16* __restrict__ Wre,  // [D_H, D_IN]
    const bf16* __restrict__ Wim,  // [D_H, D_IN]
    float* __restrict__ BuRe,      // [M, D_H]
    float* __restrict__ BuIm) {
  __shared__ __align__(16) bf16 As [2][128 * LDT];
  __shared__ __align__(16) bf16 WsR[2][128 * LDT];
  __shared__ __align__(16) bf16 WsI[2][128 * LDT];

  const int tid  = threadIdx.x;
  const int lane = tid & 31;
  const int wave = tid >> 5;
  const int wm   = wave >> 1;  // 0..3 -> M offset wm*32
  const int wn   = wave & 1;   // 0..1 -> N offset wn*64
  const int row0 = blockIdx.x * 128;
  const int col0 = blockIdx.y * 128;

  floatx8 accR[2][4] = {};
  floatx8 accI[2][4] = {};

  auto stage = [&](int buf, int kk) {
    { // A tile 128x32: two 16B async copies per thread
      const int r = tid >> 1;
      const int c = (tid & 1) * 16;
      const bf16* g = U + (size_t)(row0 + r) * D_IN + kk + c;
      bf16* l = As[buf] + r * LDT + c;
      async_b128(l,     g);
      async_b128(l + 8, g + 8);
    }
    { // both weight tiles 128x32; one row (64B) per thread, half threads per matrix
      const int t2 = tid & 127;
      const bf16* g = ((tid < 128) ? Wre : Wim) + (size_t)(col0 + t2) * D_IN + kk;
      bf16* l = ((tid < 128) ? WsR[buf] : WsI[buf]) + t2 * LDT;
      async_b128(l,      g);
      async_b128(l + 8,  g + 8);
      async_b128(l + 16, g + 16);
      async_b128(l + 24, g + 24);
    }
  };

  constexpr int NIT = D_IN / 32;
  stage(0, 0);
  for (int it = 0; it < NIT; ++it) {
    const int buf = it & 1;
    wait_async0();        // this thread's copies for buf are the only outstanding ones
    __syncthreads();      // everyone's tile landed; also fences buffer reuse (distance 2)
    if (it + 1 < NIT) stage(buf ^ 1, (it + 1) * 32);   // DMA next tile under compute

    bf16x16 a0 = frag_a(As[buf] + (wm * 32 + 0)  * LDT, lane);
    bf16x16 a1 = frag_a(As[buf] + (wm * 32 + 16) * LDT, lane);
#pragma unroll
    for (int s = 0; s < 4; ++s) {
      bf16x16 br = frag_b(WsR[buf] + (wn * 64 + s * 16) * LDT, lane);
      accR[0][s] = __builtin_amdgcn_wmma_f32_16x16x32_bf16(false, a0, false, br, (short)0, accR[0][s], false, false);
      accR[1][s] = __builtin_amdgcn_wmma_f32_16x16x32_bf16(false, a1, false, br, (short)0, accR[1][s], false, false);
      bf16x16 bi = frag_b(WsI[buf] + (wn * 64 + s * 16) * LDT, lane);
      accI[0][s] = __builtin_amdgcn_wmma_f32_16x16x32_bf16(false, a0, false, bi, (short)0, accI[0][s], false, false);
      accI[1][s] = __builtin_amdgcn_wmma_f32_16x16x32_bf16(false, a1, false, bi, (short)0, accI[1][s], false, false);
    }
  }

  // store: VGPR r -> M = r + 8*hi (ISA C/D layout), lane&15 -> N
  const int nbase = col0 + wn * 64 + (lane & 15);
  const int mbase = row0 + wm * 32 + (lane >> 4) * 8;
#pragma unroll
  for (int a = 0; a < 2; ++a) {
#pragma unroll
    for (int s = 0; s < 4; ++s) {
      float* pR = BuRe + (size_t)(mbase + a * 16) * D_H + nbase + s * 16;
      float* pI = BuIm + (size_t)(mbase + a * 16) * D_H + nbase + s * 16;
#pragma unroll
      for (int r = 0; r < 8; ++r) {
        pR[(size_t)r * D_H] = accR[a][s][r];
        pI[(size_t)r * D_H] = accI[a][s][r];
      }
    }
  }
}

// ---------------- scan: h_t = lam * h_{t-1} + Bu_t (fp32 state) ----------------

__global__ __launch_bounds__(256) void lru_scan(
    const float* __restrict__ BuRe, const float* __restrict__ BuIm,
    const float* __restrict__ lamRe, const float* __restrict__ lamIm,
    bf16* __restrict__ HsRe, bf16* __restrict__ HsIm) {
  const int h = blockIdx.x * 256 + threadIdx.x; // 0..1023, lanes span h -> coalesced
  const int b = blockIdx.y;                     // 0..7
  const float lr = lamRe[h];
  const float li = lamIm[h];
  float hr = 0.f, hi = 0.f;
  size_t idx = (size_t)b * SEQ * D_H + h;
#pragma unroll 4
  for (int t = 0; t < SEQ; ++t) {
    const float br = BuRe[idx];
    const float bi = BuIm[idx];
    const float nr = fmaf(lr, hr, fmaf(-li, hi, br));
    const float ni = fmaf(lr, hi, fmaf(li, hr, bi));
    hr = nr; hi = ni;
    HsRe[idx] = (bf16)hr;
    HsIm[idx] = (bf16)hi;
    idx += D_H;
  }
}

// ---------------- GEMM 2: y = [hs_re | hs_im | u] @ [C_re; -C_im; D]^T ----------------

__global__ __launch_bounds__(256) void lru_gemm_out(
    const bf16* __restrict__ HsRe,  // [M, D_H]
    const bf16* __restrict__ HsIm,  // [M, D_H]
    const bf16* __restrict__ Ub,    // [M, D_IN]
    const bf16* __restrict__ Wcat,  // [D_OUT, KCAT]
    float* __restrict__ Y) {        // [M, D_OUT]
  __shared__ __align__(16) bf16 As[2][128 * LDT];
  __shared__ __align__(16) bf16 Ws[2][128 * LDT];

  const int tid  = threadIdx.x;
  const int lane = tid & 31;
  const int wave = tid >> 5;
  const int wm   = wave >> 1;
  const int wn   = wave & 1;
  const int row0 = blockIdx.x * 128;
  const int col0 = blockIdx.y * 128;

  floatx8 acc[2][4] = {};

  auto stage = [&](int buf, int kk) {
    // pick the A segment for this 32-wide K tile (boundaries are 32-aligned)
    const bf16* src; int lda, kloc;
    if (kk < D_H)            { src = HsRe; lda = D_H;  kloc = kk; }
    else if (kk < 2 * D_H)   { src = HsIm; lda = D_H;  kloc = kk - D_H; }
    else                     { src = Ub;   lda = D_IN; kloc = kk - 2 * D_H; }
    { // A tile 128x32
      const int r = tid >> 1;
      const int c = (tid & 1) * 16;
      const bf16* g = src + (size_t)(row0 + r) * lda + kloc + c;
      bf16* l = As[buf] + r * LDT + c;
      async_b128(l,     g);
      async_b128(l + 8, g + 8);
    }
    { // W tile 128x32
      const int n = tid >> 1;
      const int c = (tid & 1) * 16;
      const bf16* g = Wcat + (size_t)(col0 + n) * KCAT + kk + c;
      bf16* l = Ws[buf] + n * LDT + c;
      async_b128(l,     g);
      async_b128(l + 8, g + 8);
    }
  };

  constexpr int NIT = KCAT / 32;
  stage(0, 0);
  for (int it = 0; it < NIT; ++it) {
    const int buf = it & 1;
    wait_async0();
    __syncthreads();
    if (it + 1 < NIT) stage(buf ^ 1, (it + 1) * 32);

    bf16x16 a0 = frag_a(As[buf] + (wm * 32 + 0)  * LDT, lane);
    bf16x16 a1 = frag_a(As[buf] + (wm * 32 + 16) * LDT, lane);
#pragma unroll
    for (int s = 0; s < 4; ++s) {
      bf16x16 bw = frag_b(Ws[buf] + (wn * 64 + s * 16) * LDT, lane);
      acc[0][s] = __builtin_amdgcn_wmma_f32_16x16x32_bf16(false, a0, false, bw, (short)0, acc[0][s], false, false);
      acc[1][s] = __builtin_amdgcn_wmma_f32_16x16x32_bf16(false, a1, false, bw, (short)0, acc[1][s], false, false);
    }
  }

  const int nbase = col0 + wn * 64 + (lane & 15);
  const int mbase = row0 + wm * 32 + (lane >> 4) * 8;
#pragma unroll
  for (int a = 0; a < 2; ++a) {
#pragma unroll
    for (int s = 0; s < 4; ++s) {
      float* pY = Y + (size_t)(mbase + a * 16) * D_OUT + nbase + s * 16;
#pragma unroll
      for (int r = 0; r < 8; ++r) {
        pY[(size_t)r * D_OUT] = acc[a][s][r];
      }
    }
  }
}

// ---------------- launch ----------------

extern "C" void kernel_launch(void* const* d_in, const int* in_sizes, int n_in,
                              void* d_out, int out_size, void* d_ws, size_t ws_size,
                              hipStream_t stream) {
  (void)in_sizes; (void)n_in; (void)out_size; (void)ws_size;
  const float* u         = (const float*)d_in[0];
  const float* theta_log = (const float*)d_in[1];
  const float* nu_log    = (const float*)d_in[2];
  const float* gamma_log = (const float*)d_in[3];
  const float* B_re      = (const float*)d_in[4];
  const float* B_im      = (const float*)d_in[5];
  const float* C_re      = (const float*)d_in[6];
  const float* C_im      = (const float*)d_in[7];
  const float* Dm        = (const float*)d_in[8];
  float* y = (float*)d_out;

  char* ws = (char*)d_ws;
  size_t off = 0;
  auto carve = [&](size_t bytes) -> void* {
    void* p = ws + off;
    off = (off + bytes + 255) & ~(size_t)255;
    return p;
  };
  bf16*  Ub    = (bf16*) carve((size_t)M * D_IN * 2);      // 16 MB
  bf16*  Wre   = (bf16*) carve((size_t)D_H * D_IN * 2);    // 1 MB
  bf16*  Wim   = (bf16*) carve((size_t)D_H * D_IN * 2);    // 1 MB
  bf16*  Wcat  = (bf16*) carve((size_t)D_OUT * KCAT * 2);  // 2.5 MB
  float* lamRe = (float*)carve((size_t)D_H * 4);
  float* lamIm = (float*)carve((size_t)D_H * 4);
  float* BuRe  = (float*)carve((size_t)M * D_H * 4);       // 64 MB
  float* BuIm  = (float*)carve((size_t)M * D_H * 4);       // 64 MB
  bf16*  HsRe  = (bf16*) carve((size_t)M * D_H * 2);       // 32 MB
  bf16*  HsIm  = (bf16*) carve((size_t)M * D_H * 2);       // 32 MB

  lru_prep_params<<<dim3((D_H + 255) / 256), dim3(256), 0, stream>>>(theta_log, nu_log, lamRe, lamIm);
  lru_prep_b     <<<dim3((D_H * D_IN) / 256), dim3(256), 0, stream>>>(gamma_log, B_re, B_im, Wre, Wim);
  lru_prep_u     <<<dim3((M * D_IN) / 256), dim3(256), 0, stream>>>(u, Ub);
  lru_prep_wcat  <<<dim3((D_OUT * KCAT) / 256), dim3(256), 0, stream>>>(C_re, C_im, Dm, Wcat);

  lru_gemm_bu <<<dim3(M / 128, D_H / 128), dim3(256), 0, stream>>>(Ub, Wre, Wim, BuRe, BuIm);
  lru_scan    <<<dim3(D_H / 256, NBATCH), dim3(256), 0, stream>>>(BuRe, BuIm, lamRe, lamIm, HsRe, HsIm);
  lru_gemm_out<<<dim3(M / 128, D_OUT / 128), dim3(256), 0, stream>>>(HsRe, HsIm, Ub, Wcat, y);
}
